// AttentionBlock_7198365188468
// MI455X (gfx1250) — matrix operand order repaired and checked by reading, not compiled, not run on hardware
//
#include <hip/hip_runtime.h>

// ---------------------------------------------------------------------------
// Types
// ---------------------------------------------------------------------------
typedef __attribute__((ext_vector_type(16))) __bf16 bf16x16;
typedef __attribute__((ext_vector_type(8)))  float  f32x8;
typedef __attribute__((ext_vector_type(4)))  unsigned int u32x4;

struct U32x8 { u32x4 lo, hi; };

__device__ inline bf16x16 frag_from(u32x4 lo, u32x4 hi) {
  U32x8 t; t.lo = lo; t.hi = hi;
  return __builtin_bit_cast(bf16x16, t);
}

// A-fragment (16x32 bf16): lane holds row (l&15); element j -> k = j+16*(l>>4)
// => two contiguous 16B chunks at k = 16*(l>>4) and k = 8+16*(l>>4)
__device__ inline bf16x16 loadA_frag(const unsigned short* base, int lane) {
  const unsigned short* p = base + ((lane >> 4) << 4);
  u32x4 lo = *(const u32x4*)p;
  u32x4 hi = *(const u32x4*)(p + 8);
  return frag_from(lo, hi);
}

// B-fragment (32x16 bf16): lane holds col (l&15); 16 contiguous k at 16*(l>>4)
// => a single contiguous 32B chunk (caller bakes in the 16*(l>>4) offset)
__device__ inline bf16x16 loadB_frag(const unsigned short* p) {
  return *(const bf16x16*)p;
}

__device__ inline unsigned short f2bf(float f) {
  unsigned u = __builtin_bit_cast(unsigned, f);
  unsigned r = u + 0x7fffu + ((u >> 16) & 1u);
  return (unsigned short)(r >> 16);
}

// CDNA5 async global->LDS DMA (ASYNCcnt-tracked, bypasses VGPRs).
// 16 bytes per lane: LDS dest address in VDST VGPR, 64-bit global addr in VGPR pair.
__device__ inline void async_b128(const unsigned short* gsrc, unsigned short* ldst) {
  unsigned laddr = (unsigned)(size_t)ldst;
  asm volatile("global_load_async_to_lds_b128 %0, %1, off"
               :: "v"(laddr), "v"(gsrc)
               : "memory");
}

#define WMMA_BF16(a, b, c) \
  __builtin_amdgcn_wmma_f32_16x16x32_bf16(false, (a), false, (b), (short)0, (c), false, false)

// ---------------------------------------------------------------------------
// Problem constants
// ---------------------------------------------------------------------------
#define T_   2048
#define H_   2048
#define NH_  32
#define NKV_ 8
#define HD_  64
#define QKV_ 3072
#define SW_  128

// ---------------------------------------------------------------------------
// Kernel 1: RMSNorm (f32 in) -> bf16 normed
// ---------------------------------------------------------------------------
__global__ __launch_bounds__(256) void rmsnorm_kernel(
    const float* __restrict__ x, const float* __restrict__ sc,
    unsigned short* __restrict__ out) {
  __shared__ float red[256];
  int t = blockIdx.x, tid = threadIdx.x;
  const float* row = x + (size_t)t * H_;
  float acc = 0.0f;
  for (int i = tid; i < H_; i += 256) { float v = row[i]; acc += v * v; }
  red[tid] = acc;
  __syncthreads();
  for (int off = 128; off > 0; off >>= 1) {
    if (tid < off) red[tid] += red[tid + off];
    __syncthreads();
  }
  float inv = rsqrtf(red[0] * (1.0f / (float)H_) + 1e-5f);
  for (int i = tid; i < H_; i += 256)
    out[(size_t)t * H_ + i] = f2bf(row[i] * inv * sc[i]);
}

// ---------------------------------------------------------------------------
// Kernel 2: f32 -> bf16 transposed weight conversion.
// in: K x N (row-major), out: N x K (so GEMM B-fragments are k-contiguous)
// ---------------------------------------------------------------------------
__global__ __launch_bounds__(256) void transpose_bf16_kernel(
    const float* __restrict__ in, unsigned short* __restrict__ out, int K, int N) {
  __shared__ float tile[32][33];
  int tx = threadIdx.x & 31, ty = threadIdx.x >> 5;  // ty = 0..7
  int nBase = blockIdx.x * 32, kBase = blockIdx.y * 32;
#pragma unroll
  for (int i = 0; i < 4; ++i) {
    int k = ty + i * 8;
    tile[k][tx] = in[(size_t)(kBase + k) * N + nBase + tx];
  }
  __syncthreads();
#pragma unroll
  for (int i = 0; i < 4; ++i) {
    int n = ty + i * 8;
    out[(size_t)(nBase + n) * K + kBase + tx] = f2bf(tile[tx][n]);
  }
}

// ---------------------------------------------------------------------------
// Kernel 3: bf16 GEMM  C[M,N] = A[M,K] * BT[N,K]^T + bias (+ residual)
// block = 256 threads = 8 waves (2x4), tile 64(M) x 128(N), staged K = 64.
// Double-buffered LDS; global->LDS staging via CDNA5 async-to-LDS DMA so the
// next tile's fetch overlaps the current tile's WMMAs (ASYNCcnt is in-order:
// waiting for <=6 outstanding ops guarantees the older stage landed).
// ---------------------------------------------------------------------------
__global__ __launch_bounds__(256) void gemm_bf16_kernel(
    const unsigned short* __restrict__ A, const unsigned short* __restrict__ BT,
    const float* __restrict__ bias, const float* __restrict__ residual,
    float* __restrict__ C, int M, int N, int K) {
  __shared__ unsigned short lsA[2][64 * 64];    // [m][k], k-stride 64
  __shared__ unsigned short lsB[2][128 * 64];   // [n][k], k-stride 64

  int tid = threadIdx.x;
  int lane = tid & 31, wid = tid >> 5;
  int nBase = blockIdx.x * 128, mBase = blockIdx.y * 64;
  int waveM = wid >> 2, waveN = wid & 3;        // 2 x 4 waves, each 32x32

  f32x8 zf = {0, 0, 0, 0, 0, 0, 0, 0};
  f32x8 acc[2][2] = {{zf, zf}, {zf, zf}};

  // 6 async DMA ops per wave per stage (2 for A, 4 for B)
  auto stage = [&](int b, int k0) {
#pragma unroll
    for (int j = 0; j < 2; ++j) {
      int e = (tid + 256 * j) * 8;
      int r = e >> 6, kk = e & 63;
      async_b128(&A[(size_t)(mBase + r) * K + k0 + kk], &lsA[b][e]);
    }
#pragma unroll
    for (int j = 0; j < 4; ++j) {
      int e = (tid + 256 * j) * 8;
      int n = e >> 6, kk = e & 63;
      async_b128(&BT[(size_t)(nBase + n) * K + k0 + kk], &lsB[b][e]);
    }
  };

  stage(0, 0);
  int nk = K >> 6;
  for (int i = 0; i < nk; ++i) {
    if (i + 1 < nk) {
      stage((i + 1) & 1, (i + 1) << 6);
      asm volatile("s_wait_asynccnt 0x6" ::: "memory");  // stage i landed
    } else {
      asm volatile("s_wait_asynccnt 0x0" ::: "memory");
    }
    __syncthreads();

    const unsigned short* bufA = lsA[i & 1];
    const unsigned short* bufB = lsB[i & 1];
#pragma unroll
    for (int kh = 0; kh < 2; ++kh) {
      int kk0 = kh * 32;
      bf16x16 a0 = loadA_frag(&bufA[(waveM * 32 + (lane & 15)) * 64 + kk0], lane);
      bf16x16 a1 = loadA_frag(&bufA[(waveM * 32 + 16 + (lane & 15)) * 64 + kk0], lane);
      bf16x16 b0 = loadB_frag(&bufB[(waveN * 32 + (lane & 15)) * 64 + kk0 + ((lane >> 4) << 4)]);
      bf16x16 b1 = loadB_frag(&bufB[(waveN * 32 + 16 + (lane & 15)) * 64 + kk0 + ((lane >> 4) << 4)]);
      acc[0][0] = WMMA_BF16(a0, b0, acc[0][0]);
      acc[0][1] = WMMA_BF16(a0, b1, acc[0][1]);
      acc[1][0] = WMMA_BF16(a1, b0, acc[1][0]);
      acc[1][1] = WMMA_BF16(a1, b1, acc[1][1]);
    }
    __syncthreads();
  }

  // C/D layout: element r -> row = r + 8*(lane>>4), col = lane&15
  int colBase = nBase + waveN * 32 + (lane & 15);
  int rowBase = mBase + waveM * 32 + ((lane >> 4) << 3);
  if (residual != nullptr) {
#pragma unroll
    for (int sm = 0; sm < 2; ++sm)
#pragma unroll
      for (int sn = 0; sn < 2; ++sn) {
        int col = colBase + sn * 16;
        float b = bias[col];
        size_t idx = (size_t)(rowBase + sm * 16) * N + col;
#pragma unroll
        for (int r = 0; r < 8; ++r) {
          C[idx] = acc[sm][sn][r] + b + residual[idx];
          idx += (size_t)N;
        }
      }
  } else {
#pragma unroll
    for (int sm = 0; sm < 2; ++sm)
#pragma unroll
      for (int sn = 0; sn < 2; ++sn) {
        int col = colBase + sn * 16;
        float b = bias[col];
        size_t idx = (size_t)(rowBase + sm * 16) * N + col;
#pragma unroll
        for (int r = 0; r < 8; ++r) {
          C[idx] = acc[sm][sn][r] + b;
          idx += (size_t)N;
        }
      }
  }
}

// ---------------------------------------------------------------------------
// Kernel 4: RoPE + split: qkv f32 -> q bf16 [t][h*64+d], k bf16 [t][g*64+d],
//           vT bf16 [g*64+d][t]
// ---------------------------------------------------------------------------
__global__ __launch_bounds__(256) void rope_kernel(
    const float* __restrict__ qkv, const float* __restrict__ cost,
    const float* __restrict__ sint, unsigned short* __restrict__ qb,
    unsigned short* __restrict__ kb, unsigned short* __restrict__ vT) {
  int t = blockIdx.x, tid = threadIdx.x;
  const float* row = qkv + (size_t)t * QKV_;
  const float* c = cost + (size_t)t * 32;
  const float* s = sint + (size_t)t * 32;
  // Q: 32 heads x 32 rotation pairs
  for (int p = tid; p < 1024; p += 256) {
    int hq = p >> 5, j = p & 31;
    float u1 = row[hq * 64 + j], u2 = row[hq * 64 + j + 32];
    float cj = c[j], sj = s[j];
    qb[(size_t)t * 2048 + hq * 64 + j]      = f2bf(u1 * cj - u2 * sj);
    qb[(size_t)t * 2048 + hq * 64 + j + 32] = f2bf(u2 * cj + u1 * sj);
  }
  // K: 8 groups x 32 pairs (exactly 256 threads)
  {
    int gg = tid >> 5, j = tid & 31;
    float u1 = row[2048 + gg * 64 + j], u2 = row[2048 + gg * 64 + j + 32];
    float cj = c[j], sj = s[j];
    kb[(size_t)t * 512 + gg * 64 + j]      = f2bf(u1 * cj - u2 * sj);
    kb[(size_t)t * 512 + gg * 64 + j + 32] = f2bf(u2 * cj + u1 * sj);
  }
  // V transposed: vT[(g*64+d)][t]
  for (int i = tid; i < 512; i += 256)
    vT[(size_t)i * T_ + t] = f2bf(row[2560 + i]);
}

// ---------------------------------------------------------------------------
// Kernel 5: sliding-window attention with sink, flash-style, WMMA.
// One wave per (head h, 16-query tile qt). block = 128 threads (4 waves).
// ---------------------------------------------------------------------------
__global__ __launch_bounds__(128) void attn_kernel(
    const unsigned short* __restrict__ qb, const unsigned short* __restrict__ kb,
    const unsigned short* __restrict__ vT, const float* __restrict__ sinks,
    unsigned short* __restrict__ attn) {
  __shared__ unsigned short lsP[4][16 * 32];  // per-wave P staging (16q x 32k)

  int lane = threadIdx.x & 31, wid = threadIdx.x >> 5;
  int work = blockIdx.x * 4 + wid;        // 0..4095
  int h = work >> 7;                      // 0..31
  int qt = work & 127;                    // 0..127
  int g = h >> 2;                         // kv group
  const float sm_scale = 0.125f;          // 1/sqrt(64)

  // Q fragments (A-layout): row = qt*16 + (lane&15), d split 0..31 / 32..63
  int qrow = qt * 16 + (lane & 15);
  const unsigned short* qp = qb + (size_t)qrow * 2048 + h * 64;
  bf16x16 aq0 = loadA_frag(qp, lane);
  bf16x16 aq1 = loadA_frag(qp + 32, lane);

  f32x8 zf = {0, 0, 0, 0, 0, 0, 0, 0};
  f32x8 o[4] = {zf, zf, zf, zf};
  float rmax[8], rsum[8];
#pragma unroll
  for (int r = 0; r < 8; ++r) { rmax[r] = -1e30f; rsum[r] = 0.0f; }

  // Window = keys [qt*16-128 .. qt*16+15] -> tiles qt-8..qt (9); pad to 5 pairs
  for (int pr = 0; pr < 5; ++pr) {
    int st0 = (qt - 9 + 2 * pr) * 16;  // first tile of pair
    int st1 = st0 + 16;

    // Scores S = Q (16x64) x K^T (64x16) per tile; B lane: col = key (l&15),
    // 16 contiguous d at 16*(l>>4) (+0 / +32 for the two K-steps).
    f32x8 sA = zf, sB = zf;
    {
      int s0 = st0 + (lane & 15); s0 = s0 < 0 ? 0 : (s0 > T_ - 1 ? T_ - 1 : s0);
      int s1 = st1 + (lane & 15); s1 = s1 < 0 ? 0 : (s1 > T_ - 1 ? T_ - 1 : s1);
      const unsigned short* p0 = kb + (size_t)s0 * 512 + g * 64 + ((lane >> 4) << 4);
      const unsigned short* p1 = kb + (size_t)s1 * 512 + g * 64 + ((lane >> 4) << 4);
      sA = WMMA_BF16(aq0, loadB_frag(p0), sA);
      sA = WMMA_BF16(aq1, loadB_frag(p0 + 32), sA);
      sB = WMMA_BF16(aq0, loadB_frag(p1), sB);
      sB = WMMA_BF16(aq1, loadB_frag(p1 + 32), sB);
    }

    // Mask + online softmax. C layout: row m = r + 8*(lane>>4), col n = lane&15
#pragma unroll
    for (int r = 0; r < 8; ++r) {
      int m = r + ((lane >> 4) << 3);
      int qp_ = qt * 16 + m;
      int kp0 = st0 + (lane & 15);
      int kp1 = st1 + (lane & 15);
      float v0 = sA[r] * sm_scale;
      float v1 = sB[r] * sm_scale;
      bool ok0 = (kp0 >= 0) && (kp0 <= qp_) && (qp_ - kp0 <= SW_);
      bool ok1 = (kp1 >= 0) && (kp1 <= qp_) && (qp_ - kp1 <= SW_);
      v0 = ok0 ? v0 : -3e38f;
      v1 = ok1 ? v1 : -3e38f;
      float tm = fmaxf(v0, v1);
      tm = fmaxf(tm, __shfl_xor(tm, 1, 32));
      tm = fmaxf(tm, __shfl_xor(tm, 2, 32));
      tm = fmaxf(tm, __shfl_xor(tm, 4, 32));
      tm = fmaxf(tm, __shfl_xor(tm, 8, 32));
      float nm = fmaxf(rmax[r], tm);
      float sc = __expf(rmax[r] - nm);
      float p0 = __expf(v0 - nm);
      float p1 = __expf(v1 - nm);
      float ps = p0 + p1;
      ps += __shfl_xor(ps, 1, 32);
      ps += __shfl_xor(ps, 2, 32);
      ps += __shfl_xor(ps, 4, 32);
      ps += __shfl_xor(ps, 8, 32);
      rsum[r] = rsum[r] * sc + ps;
      rmax[r] = nm;
      o[0][r] *= sc; o[1][r] *= sc; o[2][r] *= sc; o[3][r] *= sc;
      lsP[wid][m * 32 + (lane & 15)]      = f2bf(p0);
      lsP[wid][m * 32 + 16 + (lane & 15)] = f2bf(p1);
    }
    asm volatile("s_wait_dscnt 0x0" ::: "memory");

    // P (16x32) A-fragment from LDS, then P x V (32 x 64) across 4 n-tiles
    bf16x16 pf = loadA_frag(&lsP[wid][(lane & 15) * 32], lane);
#pragma unroll
    for (int nt = 0; nt < 4; ++nt) {
      int d = nt * 16 + (lane & 15);
      int t0 = st0 + ((lane >> 4) << 4);
      t0 = t0 < 0 ? 0 : t0;  // clamped rows have P == 0
      const unsigned short* vp = vT + (size_t)(g * 64 + d) * T_ + t0;
      o[nt] = WMMA_BF16(pf, loadB_frag(vp), o[nt]);
    }
  }

  // Finalize with sink column, store bf16 attn [t][h*64+d]
  float snk = sinks[h];
#pragma unroll
  for (int r = 0; r < 8; ++r) {
    float fm = fmaxf(rmax[r], snk);
    float denom = rsum[r] * __expf(rmax[r] - fm) + __expf(snk - fm);
    float fac = __expf(rmax[r] - fm) / denom;
    int t = qt * 16 + r + ((lane >> 4) << 3);
#pragma unroll
    for (int nt = 0; nt < 4; ++nt)
      attn[(size_t)t * 2048 + h * 64 + nt * 16 + (lane & 15)] =
          f2bf(o[nt][r] * fac);
  }
}

// ---------------------------------------------------------------------------
// Launch
// ---------------------------------------------------------------------------
extern "C" void kernel_launch(void* const* d_in, const int* in_sizes, int n_in,
                              void* d_out, int out_size, void* d_ws, size_t ws_size,
                              hipStream_t stream) {
  const float* x          = (const float*)d_in[0];
  const float* norm_scale = (const float*)d_in[1];
  const float* qkv_kernel = (const float*)d_in[2];
  const float* qkv_bias   = (const float*)d_in[3];
  const float* out_kernel = (const float*)d_in[4];
  const float* out_bias   = (const float*)d_in[5];
  const float* sinks      = (const float*)d_in[6];
  const float* cos_table  = (const float*)d_in[7];
  const float* sin_table  = (const float*)d_in[8];
  float* out = (float*)d_out;

  char* ws = (char*)d_ws;
  const size_t OFF_NORM = 0;                                    // 2048x2048 bf16
  const size_t OFF_WQKV = OFF_NORM + (size_t)T_ * H_ * 2;       // 3072x2048 bf16 (transposed)
  const size_t OFF_WOUT = OFF_WQKV + (size_t)H_ * QKV_ * 2;     // 2048x2048 bf16 (transposed)
  const size_t OFF_QKV  = OFF_WOUT + (size_t)H_ * H_ * 2;       // 2048x3072 f32
  const size_t OFF_Q    = OFF_QKV + (size_t)T_ * QKV_ * 4;      // 2048x2048 bf16
  const size_t OFF_K    = OFF_Q + (size_t)T_ * NH_ * HD_ * 2;   // 2048x512 bf16
  const size_t OFF_VT   = OFF_K + (size_t)T_ * NKV_ * HD_ * 2;  // 512x2048 bf16
  const size_t OFF_ATTN = OFF_VT + (size_t)NKV_ * HD_ * T_ * 2; // 2048x2048 bf16

  unsigned short* normBf = (unsigned short*)(ws + OFF_NORM);
  unsigned short* wqkvBf = (unsigned short*)(ws + OFF_WQKV);
  unsigned short* woutBf = (unsigned short*)(ws + OFF_WOUT);
  float*          qkvF   = (float*)(ws + OFF_QKV);
  unsigned short* qBf    = (unsigned short*)(ws + OFF_Q);
  unsigned short* kBf    = (unsigned short*)(ws + OFF_K);
  unsigned short* vTBf   = (unsigned short*)(ws + OFF_VT);
  unsigned short* attnBf = (unsigned short*)(ws + OFF_ATTN);

  // 1. RMSNorm -> bf16
  rmsnorm_kernel<<<T_, 256, 0, stream>>>(x, norm_scale, normBf);

  // 2. Weight conversions with transpose (W[K,N] -> WT[N,K] bf16)
  transpose_bf16_kernel<<<dim3(QKV_ / 32, H_ / 32), 256, 0, stream>>>(
      qkv_kernel, wqkvBf, H_, QKV_);
  transpose_bf16_kernel<<<dim3(H_ / 32, H_ / 32), 256, 0, stream>>>(
      out_kernel, woutBf, H_, H_);

  // 3. QKV GEMM: (2048x2048) x (2048x3072) + bias -> f32
  gemm_bf16_kernel<<<dim3(QKV_ / 128, T_ / 64), 256, 0, stream>>>(
      normBf, wqkvBf, qkv_bias, nullptr, qkvF, T_, QKV_, H_);

  // 4. RoPE + split to bf16 (V transposed)
  rope_kernel<<<T_, 256, 0, stream>>>(qkvF, cos_table, sin_table, qBf, kBf, vTBf);

  // 5. Attention: 32 heads x 128 query tiles, 4 waves/block
  attn_kernel<<<(NH_ * (T_ / 16)) / 4, 128, 0, stream>>>(qBf, kBf, vTBf, sinks,
                                                         attnBf);

  // 6. Output projection + residual: (2048x2048) x (2048x2048) + bias + x
  gemm_bf16_kernel<<<dim3(H_ / 128, T_ / 64), 256, 0, stream>>>(
      attnBf, woutBf, out_bias, x, out, T_, H_, H_);
}
